// SimpleFinanceNet_5179730559053
// MI455X (gfx1250) — matrix operand order repaired and checked by reading, not compiled, not run on hardware
//
#include <hip/hip_runtime.h>

#define T_STEPS 512
#define BATCH   2048
#define NF      64
#define NH      32
#define NA      3

typedef __attribute__((ext_vector_type(16))) _Float16 v16h;
typedef __attribute__((ext_vector_type(4)))  _Float16 v4h;
typedef __attribute__((ext_vector_type(8)))  float    v8f;

__device__ __forceinline__ float fast_tanh(float x) {
#if __has_builtin(__builtin_amdgcn_tanhf)
    return __builtin_amdgcn_tanhf(x);          // v_tanh_f32 (confirmed on gfx1250)
#elif __has_builtin(__builtin_amdgcn_exp2f) && __has_builtin(__builtin_amdgcn_rcpf)
    float e = __builtin_amdgcn_exp2f(x * 2.8853900817779268f);
    return 1.0f - 2.0f * __builtin_amdgcn_rcpf(e + 1.0f);
#else
    return tanhf(x);
#endif
}

#if __has_builtin(__builtin_amdgcn_s_barrier_signal) && \
    __has_builtin(__builtin_amdgcn_s_barrier_wait)   && \
    __has_builtin(__builtin_amdgcn_s_wait_dscnt)
#define USE_SPLIT_BARRIER 1
#else
#define USE_SPLIT_BARRIER 0
#endif

// One workgroup = 16 batch rows, 8 waves. Wave wv owns gate g = wv>>1
// (0=i 1=f 2=g 3=o) and n-tile nt = wv&1 -> gate columns [32g+16nt, +16).
__global__ __launch_bounds__(256)
void lstm_policy_kernel(const float* __restrict__ x,
                        const float* __restrict__ W_ih,
                        const float* __restrict__ W_hh,
                        const float* __restrict__ b_ih,
                        const float* __restrict__ b_hh,
                        const float* __restrict__ W_d,
                        const float* __restrict__ b_d,
                        float* __restrict__ out) {
    // h (f16) in WMMA A-fragment layout: hbuf[lane][e] = h[m=lane&15][k],
    // k = e + 8*(lane>>4) + (e>=8 ? 8 : 0)
    __shared__ __align__(32) _Float16 hbuf[32][16];
    // double-buffered x tile, pre-converted to f16, A-fragment layout per k-block
    __shared__ __align__(32) _Float16 xbuf[2][2][32][16];
    // activated gates, AoS over the 4 gates -> one float4 read per (m,n)
    __shared__ float gbuf[16][32][4];

    const int tid  = threadIdx.x;
    const int wv   = tid >> 5;
    const int g    = wv >> 1;           // gate id
    const int nt   = wv & 1;            // n-tile within gate
    const int lane = tid & 31;
    const int hl   = lane >> 4;
    const int ln   = lane & 15;
    const int row0 = blockIdx.x * 16;

    // ---- weight B-fragments (resident): elem e -> K = e + 8*hl + (e>=8 ? 8 : 0), N = ln ----
    const int gcol = 32 * g + 16 * nt + ln;
    v16h wxB[2], whB;
    {
        const float* wr = W_ih + gcol * NF;
        #pragma unroll
        for (int kb = 0; kb < 2; ++kb) {
            const float* p = wr + 32 * kb + 8 * hl;
            #pragma unroll
            for (int q = 0; q < 16; ++q) wxB[kb][q] = (_Float16)p[q + ((q >= 8) ? 8 : 0)];
        }
        const float* hr = W_hh + gcol * NH + 8 * hl;
        #pragma unroll
        for (int q = 0; q < 16; ++q) whB[q] = (_Float16)hr[q + ((q >= 8) ? 8 : 0)];
    }
    const float bias = b_ih[gcol] + b_hh[gcol];
    // branchless activation: act(v) = actA * tanh(actB*v) + actC
    const float actA = (g == 2) ? 1.0f : 0.5f;
    const float actB = (g == 2) ? 1.0f : 0.5f;
    const float actC = (g == 2) ? 0.0f : 0.5f;

    // ---- x staging: thread -> 4 consecutive floats of the 16x64 tile -------
    const int sm  = tid >> 4;           // tile row 0..15
    const int sk  = (tid & 15) * 4;     // k 0..60
    const int skb = sk >> 5;
    const int skk = sk & 31;
    const int shl = (skk >> 3) & 1;
    const int se  = (skk & 7) + ((skk >= 16) ? 8 : 0);
    const float* xg = x + ((size_t)row0 + sm) * NF + sk;
    v4h* xdst0 = (v4h*)&xbuf[0][skb][sm + 16 * shl][se];
    v4h* xdst1 = (v4h*)&xbuf[1][skb][sm + 16 * shl][se];

    // ---- init state + preload x tile for t = 0 -----------------------------
    #pragma unroll
    for (int j = 0; j < 2; ++j) {
        int u = tid + 256 * j;
        hbuf[u >> 4][u & 15] = (_Float16)0.0f;
    }
    {
        float4 v = *(const float4*)xg;
        v4h pk; pk[0] = (_Float16)v.x; pk[1] = (_Float16)v.y;
                pk[2] = (_Float16)v.z; pk[3] = (_Float16)v.w;
        *xdst0 = pk;
    }
    __syncthreads();

    float creg[2] = {0.0f, 0.0f};       // private cell state (same (m,n) each step)
    int cur = 0;
    for (int t = 0; t < T_STEPS; ++t) {
        // issue next step's x load early: latency hides under the whole step
        const int tn = (t + 1 < T_STEPS) ? (t + 1) : t;
        float4 vnxt = *(const float4*)(xg + (size_t)tn * (BATCH * NF));

        // ---- fragments from LDS ----
        v16h xa0 = *(const v16h*)(&xbuf[cur][0][lane][0]);
        v16h xa1 = *(const v16h*)(&xbuf[cur][1][lane][0]);
        v16h ha  = *(const v16h*)(&hbuf[lane][0]);

        // ---- gates = x @ W_ih^T + h @ W_hh^T + bias: three independent -----
        // WMMA chains -> zero WMMA->WMMA RAW hazards; merged with packed adds.
        v8f accA, accB, accC;
        #pragma unroll
        for (int r = 0; r < 8; ++r) { accA[r] = bias; accB[r] = 0.0f; accC[r] = 0.0f; }
        accA = __builtin_amdgcn_wmma_f32_16x16x32_f16(false, xa0, false, wxB[0], (short)0, accA, false, false);
        accB = __builtin_amdgcn_wmma_f32_16x16x32_f16(false, xa1, false, wxB[1], (short)0, accB, false, false);
        accC = __builtin_amdgcn_wmma_f32_16x16x32_f16(false, ha,  false, whB,    (short)0, accC, false, false);

        // ---- branchless per-gate activation --------------------------------
        #pragma unroll
        for (int r = 0; r < 8; ++r)
            accA[r] = fmaf(actA, fast_tanh(actB * ((accA[r] + accB[r]) + accC[r])), actC);

        // ---- publish gates: C/D elem r -> M = r + 8*hl, N = 16*nt + ln -----
        #pragma unroll
        for (int r = 0; r < 8; ++r)
            gbuf[r + 8 * hl][16 * nt + ln][g] = accA[r];

#if USE_SPLIT_BARRIER
        __builtin_amdgcn_s_wait_dscnt(0);       // gate stores committed
        __builtin_amdgcn_s_barrier_signal(-1);  // arrive at barrier 1
#endif
        // ---- stage next x tile (nobody reads it until after barrier 2) -----
        {
            v4h pk; pk[0] = (_Float16)vnxt.x; pk[1] = (_Float16)vnxt.y;
                    pk[2] = (_Float16)vnxt.z; pk[3] = (_Float16)vnxt.w;
            *(cur ? xdst0 : xdst1) = pk;
        }
#if USE_SPLIT_BARRIER
        __builtin_amdgcn_s_barrier_wait(-1);
#else
        __syncthreads();
#endif

        // ---- c/h update: 512 elements over 256 threads, c in registers -----
        #pragma unroll
        for (int j = 0; j < 2; ++j) {
            int u = tid + 256 * j;
            int m = u >> 5, n = u & 31;
            const float4 gg = *(const float4*)&gbuf[m][n][0];  // i,f,g,o
            float c = fmaf(gg.y, creg[j], gg.x * gg.z);
            creg[j] = c;
            float h = gg.w * fast_tanh(c);
            // scatter h into A-fragment layout for next step
            hbuf[m + 16 * ((n >> 3) & 1)][(n & 7) + ((n >= 16) ? 8 : 0)] = (_Float16)h;
        }
#if USE_SPLIT_BARRIER
        __builtin_amdgcn_s_wait_dscnt(0);       // h stores committed
        __builtin_amdgcn_s_barrier_signal(-1);
        __builtin_amdgcn_s_barrier_wait(-1);
#else
        __syncthreads();
#endif
        cur ^= 1;
    }

    // ---- dense head: out = relu(h_T) @ W_d^T + b_d (h_T read from hbuf) ----
    if (tid < 16 * NA) {
        int m = tid / NA, a = tid % NA;
        float s = b_d[a];
        #pragma unroll
        for (int n = 0; n < NH; ++n) {
            float hv = (float)hbuf[m + 16 * ((n >> 3) & 1)][(n & 7) + ((n >= 16) ? 8 : 0)];
            s = fmaf(fmaxf(hv, 0.0f), W_d[a * NH + n], s);
        }
        out[(size_t)(row0 + m) * NA + a] = s;
    }
}

extern "C" void kernel_launch(void* const* d_in, const int* in_sizes, int n_in,
                              void* d_out, int out_size, void* d_ws, size_t ws_size,
                              hipStream_t stream) {
    const float* x    = (const float*)d_in[0];
    const float* W_ih = (const float*)d_in[1];
    const float* W_hh = (const float*)d_in[2];
    const float* b_ih = (const float*)d_in[3];
    const float* b_hh = (const float*)d_in[4];
    const float* W_d  = (const float*)d_in[5];
    const float* b_d  = (const float*)d_in[6];
    float* out = (float*)d_out;

    dim3 grid(BATCH / 16);   // 128 workgroups x 16 batch rows
    dim3 block(256);         // 8 waves: one (gate, n-tile) pair per wave
    hipLaunchKernelGGL(lstm_policy_kernel, grid, block, 0, stream,
                       x, W_ih, W_hh, b_ih, b_hh, W_d, b_d, out);
}
